// MultiHeadCausalAttention_34119220199728
// MI455X (gfx1250) — matrix operand order repaired and checked by reading, not compiled, not run on hardware
//
#include <hip/hip_runtime.h>
#include <hip/hip_bf16.h>
#include <stdint.h>

// ---------------------------------------------------------------------------
// MultiHeadCausalAttention for MI455X (gfx1250, wave32, WMMA)
//   B=4, T=2048, C=1024, H=16, Dh=64
// Pipeline:
//   [cvt]   x, W_qkv, W_out  : f32 -> bf16 (workspace)
//   [gemm]  qkv = x @ W_qkv        (bf16 in, f32 acc via v_wmma, bf16 out)
//   [flash] O   = softmax(QK^T)V   (fused, causal, online softmax, bf16 out)
//   [gemm]  out = O @ W_out        (f32 out to d_out)
// GEMM uses double-buffered LDS tiles staged with GLOBAL_LOAD_ASYNC_TO_LDS
// (ASYNCcnt) when available, so tile k+1 streams in while WMMAs chew tile k.
// ---------------------------------------------------------------------------

typedef __attribute__((ext_vector_type(16))) __bf16 v16bf;
typedef __attribute__((ext_vector_type(8)))  float  v8f;

#if defined(__gfx1250__) && defined(__has_builtin)
#if __has_builtin(__builtin_amdgcn_global_load_async_to_lds_b128)
#define USE_ASYNC_LDS 1
#endif
#endif
#ifndef USE_ASYNC_LDS
#define USE_ASYNC_LDS 0
#endif

#if USE_ASYNC_LDS
// Builtin signature (from hipcc diagnostic): (int4 AS(1)*, int4 AS(3)*, imm, imm)
typedef int i4v __attribute__((vector_size(16)));
typedef __attribute__((address_space(1))) i4v* g_i4p;
typedef __attribute__((address_space(3))) i4v* l_i4p;
#endif

union FragU {
    v16bf bf;
    uint4 u4[2];
};

static __device__ inline v16bf load_frag(const __bf16* p0, const __bf16* p1) {
    FragU f;
    f.u4[0] = *reinterpret_cast<const uint4*>(p0);
    f.u4[1] = *reinterpret_cast<const uint4*>(p1);
    return f.bf;
}

static __device__ inline __bf16 f32_to_bf16(float x) {
    union { float f; uint32_t u; } c; c.f = x;
    uint32_t r = (c.u + 0x7FFFu + ((c.u >> 16) & 1u)) >> 16;   // RNE
    union { unsigned short s; __bf16 b; } o; o.s = (unsigned short)r;
    return o.b;
}

static __device__ inline v8f wmma_bf16(v16bf a, v16bf b, v8f c) {
    // D(16x16,f32) = A(16x32,bf16) * B(32x16,bf16) + C
    return __builtin_amdgcn_wmma_f32_16x16x32_bf16(
        /*neg_a=*/false, a, /*neg_b=*/false, b,
        /*c_mod=*/(short)0, c, /*reuse_a=*/false, /*reuse_b=*/false);
}

// 16-byte global -> LDS copy: async (ASYNCcnt) on gfx1250, sync fallback.
static __device__ inline void copy16_g2l(void* lds, const void* g) {
#if USE_ASYNC_LDS
    __builtin_amdgcn_global_load_async_to_lds_b128((g_i4p)g, (l_i4p)lds, 0, 0);
#else
    *reinterpret_cast<uint4*>(lds) = *reinterpret_cast<const uint4*>(g);
#endif
}

// wait until at most N async copies from this wave remain in flight
static __device__ inline void wait_async_le4() {
#if USE_ASYNC_LDS
    asm volatile("s_wait_asynccnt 0x4" ::: "memory");
#endif
}
static __device__ inline void wait_async_0() {
#if USE_ASYNC_LDS
    asm volatile("s_wait_asynccnt 0x0" ::: "memory");
#endif
}

// ---------------------------------------------------------------------------
// Kernel 1: elementwise f32 -> bf16 (vectorized x4; all sizes divisible by 1024)
// ---------------------------------------------------------------------------
struct bf16x4 { __bf16 v[4]; };

__global__ __launch_bounds__(256)
void cvt_f32_bf16(const float* __restrict__ in, __bf16* __restrict__ out, int n4) {
    int i = blockIdx.x * blockDim.x + threadIdx.x;
    if (i >= n4) return;
    float4 f = reinterpret_cast<const float4*>(in)[i];
    bf16x4 o;
    o.v[0] = f32_to_bf16(f.x);
    o.v[1] = f32_to_bf16(f.y);
    o.v[2] = f32_to_bf16(f.z);
    o.v[3] = f32_to_bf16(f.w);
    reinterpret_cast<bf16x4*>(out)[i] = o;
}

// ---------------------------------------------------------------------------
// Kernel 2/4: C(MxN) = A(MxK) @ B(KxN), bf16 inputs row-major, f32 accumulate.
// Block tile 128x128, BK=32, 256 threads = 8 waves (4x2), wave tile 32x64.
// Double-buffered LDS; each thread stages 4x16B per tile (2 for A, 2 for B).
// M,N,K are multiples of the tile sizes for all uses here (no bounds checks).
// ---------------------------------------------------------------------------
template <bool OUT_F32>
__global__ __launch_bounds__(256)
void gemm_bf16_128x128(const __bf16* __restrict__ A, const __bf16* __restrict__ B,
                       void* __restrict__ Cout, int M, int N, int K) {
    __shared__ __align__(16) __bf16 As[2][128][40];   // 32 cols + 8 pad (80B rows)
    __shared__ __align__(16) __bf16 Bs[2][32][136];   // 128 cols + 8 pad (272B rows)

    const int tid  = threadIdx.x;
    const int w    = tid >> 5;
    const int lane = tid & 31;
    const int lm   = lane & 15;
    const int half = lane >> 4;
    const int wm   = w >> 1;          // 0..3
    const int wn   = w & 1;           // 0..1
    const int m0   = blockIdx.y * 128;
    const int n0   = blockIdx.x * 128;

    // staging indices (per thread: 32B of A row + 32B of B row)
    const int ar = tid >> 1;          // 0..127
    const int ac = (tid & 1) * 16;    // 0,16
    const int br = tid >> 3;          // 0..31
    const int bc = (tid & 7) * 16;    // 0..112

    const __bf16* gA = A + (size_t)(m0 + ar) * K + ac;   // advance by k0
    const __bf16* gB = B + (size_t)br * N + n0 + bc;     // advance by k0*N

    v8f acc[2][4];
#pragma unroll
    for (int m = 0; m < 2; ++m)
#pragma unroll
        for (int n = 0; n < 4; ++n)
#pragma unroll
            for (int v = 0; v < 8; ++v)
                acc[m][n][v] = 0.0f;

    // prologue: stage tile 0 into buffer 0
    copy16_g2l(&As[0][ar][ac],     gA);
    copy16_g2l(&As[0][ar][ac + 8], gA + 8);
    copy16_g2l(&Bs[0][br][bc],     gB);
    copy16_g2l(&Bs[0][br][bc + 8], gB + 8);

    for (int k0 = 0; k0 < K; k0 += 32) {
        const int buf = (k0 >> 5) & 1;
        __syncthreads();                       // everyone done with buf^1 data
        if (k0 + 32 < K) {                     // stage next tile into buf^1
            const __bf16* nA = gA + (k0 + 32);
            const __bf16* nB = gB + (size_t)(k0 + 32) * N;
            copy16_g2l(&As[buf ^ 1][ar][ac],     nA);
            copy16_g2l(&As[buf ^ 1][ar][ac + 8], nA + 8);
            copy16_g2l(&Bs[buf ^ 1][br][bc],     nB);
            copy16_g2l(&Bs[buf ^ 1][br][bc + 8], nB + 8);
            wait_async_le4();                  // buf's 4 copies done; 4 in flight
        } else {
            wait_async_0();
        }
        __syncthreads();                       // buf fully populated block-wide

        // A fragments: lane holds row M=lm, K offsets {8*half..}, {16+8*half..}
        v16bf af[2];
#pragma unroll
        for (int m = 0; m < 2; ++m) {
            const __bf16* rowp = &As[buf][wm * 32 + 16 * m + lm][0];
            af[m] = load_frag(rowp + 8 * half, rowp + 16 + 8 * half);
        }
#pragma unroll
        for (int n = 0; n < 4; ++n) {
            // B fragment: lane holds column N=lm, K offsets {8*half..}, {16+8*half..}
            v16bf bfrag;
#pragma unroll
            for (int e = 0; e < 16; ++e)
                bfrag[e] = Bs[buf][(e < 8 ? 8 * half + e : 8 + 8 * half + e)][wn * 64 + n * 16 + lm];
#pragma unroll
            for (int m = 0; m < 2; ++m)
                acc[m][n] = wmma_bf16(af[m], bfrag, acc[m][n]);
        }
    }

    // branch-free epilogue: C/D layout -> lane column N=lm, row = v + 8*half
    const int row0 = m0 + wm * 32 + 8 * half;
    const int col0 = n0 + wn * 64 + lm;
#pragma unroll
    for (int m = 0; m < 2; ++m)
#pragma unroll
        for (int v = 0; v < 8; ++v) {
            const size_t rbase = (size_t)(row0 + 16 * m + v) * N + col0;
            if (OUT_F32) {
                float* Cf = (float*)Cout;
#pragma unroll
                for (int n = 0; n < 4; ++n)
                    Cf[rbase + 16 * n] = acc[m][n][v];
            } else {
                __bf16* Cb = (__bf16*)Cout;
#pragma unroll
                for (int n = 0; n < 4; ++n)
                    Cb[rbase + 16 * n] = f32_to_bf16(acc[m][n][v]);
            }
        }
}

// ---------------------------------------------------------------------------
// Kernel 3: fused causal flash attention.
//   qkv: (B*T, 3072) bf16 row-major; Q at col 0, K at 1024, V at 2048, +h*64.
//   O:   (B*T, 1024) bf16 row-major (head h occupies cols h*64..h*64+63).
// Grid: (T/128, H, B); 256 threads = 8 waves; wave owns 16 query rows.
// KV tile = 64 rows. K stored transposed in LDS (KT[d][kv]) ; the WMMA
// B-operand is gathered as 16 strided 2-byte reads per fragment. V staged via async-to-LDS.
// Online softmax in log2 domain.
// ---------------------------------------------------------------------------
#define ATT_T  2048
#define ATT_3C 3072
#define ATT_C  1024

__global__ __launch_bounds__(256)
void mha_flash_fwd(const __bf16* __restrict__ qkv, __bf16* __restrict__ O) {
    __shared__ __align__(16) __bf16 KT[64][72];      // [d][kv], padded
    __shared__ __align__(16) __bf16 Vs[64][72];      // [kv][d], padded
    __shared__ __align__(16) __bf16 Ps[8][16][72];   // per-wave P scratch

    const int tid  = threadIdx.x;
    const int w    = tid >> 5;
    const int lane = tid & 31;
    const int lm   = lane & 15;
    const int half = lane >> 4;

    const int b  = blockIdx.z;
    const int h  = blockIdx.y;
    const int q0 = blockIdx.x * 128;

    // Load Q fragments (A-operand, 16 rows x 64 d, kept resident)
    const int qrow = q0 + w * 16 + lm;
    const __bf16* qptr = qkv + ((size_t)(b * ATT_T) + qrow) * ATT_3C + h * 64;
    v16bf qf[2];
#pragma unroll
    for (int f = 0; f < 2; ++f)
        qf[f] = load_frag(qptr + 32 * f + 8 * half, qptr + 32 * f + 16 + 8 * half);

    v8f oacc[4];
#pragma unroll
    for (int n = 0; n < 4; ++n)
#pragma unroll
        for (int v = 0; v < 8; ++v) oacc[n][v] = 0.0f;

    float mrow[8], lrow[8];
#pragma unroll
    for (int v = 0; v < 8; ++v) { mrow[v] = -1e30f; lrow[v] = 0.0f; }

    const float sscale = 0.125f * 1.44269504088896f;   // 1/sqrt(64) * log2(e)

    const int kv_lim = q0 + 128;                        // causal upper bound
    for (int kv0 = 0; kv0 < kv_lim; kv0 += 64) {
        __syncthreads();
        // ---- stage K (transposed, sync) and V (async) tiles -----------------
        {
            const int r  = tid >> 2;            // kv row within tile, 0..63
            const int cs = (tid & 3) * 16;      // d segment
            const __bf16* kp = qkv + ((size_t)(b * ATT_T) + kv0 + r) * ATT_3C
                                   + 1024 + h * 64 + cs;
            const __bf16* vp = kp + 1024;       // V lives 1024 cols past K
            copy16_g2l(&Vs[r][cs],     vp);
            copy16_g2l(&Vs[r][cs + 8], vp + 8);
            FragU kf;
            kf.u4[0] = reinterpret_cast<const uint4*>(kp)[0];
            kf.u4[1] = reinterpret_cast<const uint4*>(kp)[1];
#pragma unroll
            for (int e = 0; e < 16; ++e)
                KT[cs + e][r] = kf.bf[e];
            wait_async_0();
        }
        __syncthreads();

        // ---- S = Q @ K^T (16 x 64 per wave) ---------------------------------
        v8f s[4];
#pragma unroll
        for (int n = 0; n < 4; ++n)
#pragma unroll
            for (int v = 0; v < 8; ++v) s[n][v] = 0.0f;

#pragma unroll
        for (int ss = 0; ss < 2; ++ss) {
#pragma unroll
            for (int n = 0; n < 4; ++n) {
                v16bf bfrag;
#pragma unroll
                for (int e = 0; e < 16; ++e)
                    bfrag[e] = KT[32 * ss + (e < 8 ? 8 * half + e : 8 + 8 * half + e)][n * 16 + lm];
                s[n] = wmma_bf16(qf[ss], bfrag, s[n]);
            }
        }

        // ---- causal mask + online softmax (log2 domain) ---------------------
        float p[4][8];
#pragma unroll
        for (int v = 0; v < 8; ++v) {
            const int rq = q0 + w * 16 + v + 8 * half;   // this row's query idx
            float mx = -1e30f;
#pragma unroll
            for (int n = 0; n < 4; ++n) {
                float sv = s[n][v] * sscale;
                if (kv0 + n * 16 + lm > rq) sv = -1e30f;
                p[n][v] = sv;
                mx = fmaxf(mx, sv);
            }
            // row max across the 16 lanes holding this row
#pragma unroll
            for (int msk = 1; msk < 16; msk <<= 1)
                mx = fmaxf(mx, __shfl_xor(mx, msk, 16));
            float mnew  = fmaxf(mrow[v], mx);
            float alpha = exp2f(mrow[v] - mnew);
            float rs = 0.0f;
#pragma unroll
            for (int n = 0; n < 4; ++n) {
                p[n][v] = exp2f(p[n][v] - mnew);
                rs += p[n][v];
            }
#pragma unroll
            for (int msk = 1; msk < 16; msk <<= 1)
                rs += __shfl_xor(rs, msk, 16);
            lrow[v] = lrow[v] * alpha + rs;
            mrow[v] = mnew;
#pragma unroll
            for (int nd = 0; nd < 4; ++nd)
                oacc[nd][v] *= alpha;
        }

        // ---- P: C/D layout -> A layout via per-wave LDS scratch -------------
#pragma unroll
        for (int n = 0; n < 4; ++n)
#pragma unroll
            for (int v = 0; v < 8; ++v)
                Ps[w][v + 8 * half][n * 16 + lm] = f32_to_bf16(p[n][v]);
        // same-wave LDS RAW: DS ops are in-order per wave; compiler waits DScnt

        // ---- O += P @ V ------------------------------------------------------
#pragma unroll
        for (int ss = 0; ss < 2; ++ss) {
            const __bf16* prow = &Ps[w][lm][0];
            v16bf pf = load_frag(prow + 32 * ss + 8 * half,
                                 prow + 32 * ss + 16 + 8 * half);
#pragma unroll
            for (int nd = 0; nd < 4; ++nd) {
                v16bf vfrag;
#pragma unroll
                for (int e = 0; e < 16; ++e)
                    vfrag[e] = Vs[32 * ss + (e < 8 ? 8 * half + e : 8 + 8 * half + e)][nd * 16 + lm];
                oacc[nd] = wmma_bf16(pf, vfrag, oacc[nd]);
            }
        }
    }

    // ---- normalize and write O ---------------------------------------------
#pragma unroll
    for (int nd = 0; nd < 4; ++nd)
#pragma unroll
        for (int v = 0; v < 8; ++v) {
            float o = oacc[nd][v] / lrow[v];
            int t   = q0 + w * 16 + v + 8 * half;
            int col = nd * 16 + lm;
            O[((size_t)(b * ATT_T) + t) * ATT_C + h * 64 + col] = f32_to_bf16(o);
        }
}

// ---------------------------------------------------------------------------
// Host launcher
// ---------------------------------------------------------------------------
extern "C" void kernel_launch(void* const* d_in, const int* in_sizes, int n_in,
                              void* d_out, int out_size, void* d_ws, size_t ws_size,
                              hipStream_t stream) {
    (void)in_sizes; (void)n_in; (void)out_size; (void)ws_size;

    const float* x     = (const float*)d_in[0];   // (4,2048,1024)
    const float* W_qkv = (const float*)d_in[1];   // (1024,3072)
    const float* W_out = (const float*)d_in[2];   // (1024,1024)
    float* out = (float*)d_out;                   // (4,2048,1024)

    const size_t xN    = (size_t)4 * 2048 * 1024;     // 8,388,608
    const size_t WqN   = (size_t)1024 * 3072;         // 3,145,728
    const size_t WoN   = (size_t)1024 * 1024;         // 1,048,576
    const size_t qkvN  = (size_t)8192 * 3072;         // 25,165,824

    char* ws = (char*)d_ws;
    __bf16* xb    = (__bf16*)(ws);
    __bf16* Wqb   = (__bf16*)(ws + 2 * xN);
    __bf16* Wob   = (__bf16*)(ws + 2 * (xN + WqN));
    __bf16* qkvb  = (__bf16*)(ws + 2 * (xN + WqN + WoN));
    __bf16* Ob    = (__bf16*)(ws + 2 * (xN + WqN + WoN + qkvN));
    // total workspace: 92,274,688 bytes

    // 1) f32 -> bf16 conversions
    cvt_f32_bf16<<<dim3((unsigned)(xN  / 4 / 256)), 256, 0, stream>>>(x,     xb,  (int)(xN  / 4));
    cvt_f32_bf16<<<dim3((unsigned)(WqN / 4 / 256)), 256, 0, stream>>>(W_qkv, Wqb, (int)(WqN / 4));
    cvt_f32_bf16<<<dim3((unsigned)(WoN / 4 / 256)), 256, 0, stream>>>(W_out, Wob, (int)(WoN / 4));

    // 2) qkv = x @ W_qkv   (8192 x 3072 x 1024), bf16 out
    gemm_bf16_128x128<false><<<dim3(3072 / 128, 8192 / 128), 256, 0, stream>>>(
        xb, Wqb, (void*)qkvb, 8192, 3072, 1024);

    // 3) fused causal attention -> O (bf16, (B*T,1024))
    mha_flash_fwd<<<dim3(2048 / 128, 16, 4), 256, 0, stream>>>(qkvb, Ob);

    // 4) out = O @ W_out   (8192 x 1024 x 1024), f32 out
    gemm_bf16_128x128<true><<<dim3(1024 / 128, 8192 / 128), 256, 0, stream>>>(
        Ob, Wob, (void*)out, 8192, 1024, 1024);
}
